// Attention_81140522156717
// MI455X (gfx1250) — compile-verified, matrix operand order checked by reading, not run on hardware
//
#include <hip/hip_runtime.h>
#include <hip/hip_bf16.h>

// ---------------- problem constants ----------------
#define BATCH 2
#define SEQ   4096
#define EMB   768
#define HEADS 12
#define HDIM  64
#define QKV_N (3 * EMB)          // 2304
#define MTOT  (BATCH * SEQ)      // 8192
#define SCALING 0.125f           // 64^-0.5
#define LDS_STRIDE 66            // 64 bf16 + 1 DWORD TDM pad per row (132 B)

// ---------------- types ----------------
typedef __bf16 v16bf __attribute__((ext_vector_type(16)));
typedef __bf16 bf2v  __attribute__((ext_vector_type(2)));
typedef float  v8f   __attribute__((ext_vector_type(8)));
typedef unsigned int u32x4 __attribute__((ext_vector_type(4)));
typedef int          i32x4 __attribute__((ext_vector_type(4)));
typedef int          i32x8 __attribute__((ext_vector_type(8)));

#define WMMA_BF16(A, B, C) \
    __builtin_amdgcn_wmma_f32_16x16x32_bf16(false, (A), false, (B), (short)0, (C), false, false)

// sched_group_barrier masks
#define SGB_VALU      0x002
#define SGB_MFMA_WMMA 0x008
#define SGB_VMEM_READ 0x020

// CDNA5 16-bit A/B fragment K index for element e (0..15) in lane group g (= lane>>4):
// lanes 0-15 hold K = {0..7, 16..23}; lanes 16-31 hold K = {8..15, 24..31}
__device__ __forceinline__ int k_of(int e, int g) {
    return g * 8 + e + ((e >= 8) ? 8 : 0);
}

// Raw fp32 fragment: loads only (no conversion), so all global_load_b128's of a
// K-step can be grouped ahead of the converts by sched_group_barrier.
struct rawfrag {
    float2 v[8];
};

__device__ __forceinline__ void load_raw_f32(rawfrag& r, const float* __restrict__ base,
                                             int stride, int row, int kbase, int g) {
    const float* p = base + (size_t)row * stride + kbase;
#pragma unroll
    for (int e = 0; e < 16; e += 2)
        r.v[e / 2] = *(const float2*)(p + k_of(e, g));
}

__device__ __forceinline__ v16bf cvt_frag(const rawfrag& r) {
    v16bf f;
#pragma unroll
    for (int e = 0; e < 16; e += 2) {
        f[e]     = (__bf16)r.v[e / 2].x;
        f[e + 1] = (__bf16)r.v[e / 2].y;
    }
    return f;
}

__device__ __forceinline__ v16bf load_frag_bf16(const __bf16* __restrict__ base, int stride,
                                                int row, int kbase, int g) {
    v16bf f;
    const __bf16* p = base + (size_t)row * stride + kbase;
#pragma unroll
    for (int e = 0; e < 16; e += 2) {
        bf2v v = *(const bf2v*)(p + k_of(e, g));
        f[e]     = v[0];
        f[e + 1] = v[1];
    }
    return f;
}

// ---------------- Tensor Data Mover: 2D tile -> LDS ----------------
// D# per CDNA5 ISA ch.8: group0 = {count, lds_addr, global_addr, type=2};
// group1 = {flags/data_size/pad, tensor dims, tile dims (64x64), dim0 stride}.
// Padding: +1 DWORD after every 32 DWORDs (=128B row) -> LDS row stride 132 B.
// This toolchain uses the 6-arg builtin: (u32x4, i32x8, i32x4, i32x4, i32x8, i32 cpol).
__device__ __forceinline__ void tdm_load_2d(unsigned int lds_off, const __bf16* gptr,
                                            unsigned int stride0_elems) {
    unsigned long long ga = (unsigned long long)gptr;
    u32x4 g0;
    g0[0] = 1u;                                            // count=1, user descriptor
    g0[1] = lds_off;                                       // LDS byte address
    g0[2] = (unsigned int)ga;                              // global_addr[31:0]
    g0[3] = (unsigned int)((ga >> 32) & 0x1FFFFFFu) | (2u << 30);  // addr[56:32] | type=2

    i32x8 g1;
    g1[0] = (int)((1u << 16)      // data_size = 1 -> 2 bytes
                | (1u << 20)      // pad_enable
                | (4u << 22));    // pad_interval = 4 -> every 32 DWORDs (128 B)
                                  // pad_amount = 0 -> 1 DWORD
    g1[1] = (int)((64u & 0xFFFFu) << 16);                  // tensor_dim0 = 64 (bits 79:48)
    g1[2] = (int)((64u & 0xFFFFu) << 16);                  // tensor_dim1 = 64 (bits 111:80)
    g1[3] = (int)(64u << 16);                              // tile_dim0 = 64 (bits 127:112)
    g1[4] = (int)64;                                       // tile_dim1 = 64, tile_dim2 = 0
    g1[5] = (int)stride0_elems;                            // tensor_dim0_stride[31:0]
    g1[6] = 0;
    g1[7] = 0;

    i32x4 z4 = {0, 0, 0, 0};
    i32x8 z8 = {0, 0, 0, 0, 0, 0, 0, 0};
    __builtin_amdgcn_tensor_load_to_lds(g0, g1, z4, z4, z8, 0);
}

__device__ __forceinline__ unsigned int lds_off_of(const void* p) {
    // flat shared address low 32 bits == LDS byte offset (aperture mapping)
    return (unsigned int)(unsigned long long)p;
}

// =====================================================================
// Kernel 1: fused QKV projection  qkv = x @ Wqkv^T + b ; scatter to
// bf16 Q (pre-scaled) [B*H,S,D], K [B*H,S,D], V transposed [B*H,D,S].
// Block = 128 threads (4 waves). Wave tile = 16(M) x 64(N).
// =====================================================================
__global__ __launch_bounds__(128)
void qkv_proj_kernel(const float* __restrict__ x,      // [MTOT, EMB]
                     const float* __restrict__ w,      // [QKV_N, EMB]
                     const float* __restrict__ bias,   // [QKV_N]
                     __bf16* __restrict__ Q,
                     __bf16* __restrict__ K,
                     __bf16* __restrict__ VT) {
    const int lane   = threadIdx.x & 31;
    const int wave   = threadIdx.x >> 5;
    const int lane16 = lane & 15;
    const int g      = lane >> 4;

    const int mbase = blockIdx.y * 16;
    const int nwave = blockIdx.x * 256 + wave * 64;

    v8f acc[4];
#pragma unroll
    for (int t = 0; t < 4; ++t) acc[t] = (v8f)(0.0f);

    const int arow = mbase + lane16;
    for (int kb = 0; kb < EMB; kb += 32) {
        // phase 1: issue ALL loads for this k-step (A + 4 B tiles = 20 b128 loads)
        rawfrag ra, rb[4];
        load_raw_f32(ra, x, EMB, arow, kb, g);
#pragma unroll
        for (int t = 0; t < 4; ++t)
            load_raw_f32(rb[t], w, EMB, nwave + t * 16 + lane16, kb, g);
        // phase 2: convert + matrix ops
        v16bf a = cvt_frag(ra);
#pragma unroll
        for (int t = 0; t < 4; ++t)
            acc[t] = WMMA_BF16(a, cvt_frag(rb[t]), acc[t]);
        // force pipeline: loads first (one batched wait), then cvts, then WMMAs
        __builtin_amdgcn_sched_group_barrier(SGB_VMEM_READ, 20, 0);
        __builtin_amdgcn_sched_group_barrier(SGB_VALU,      40, 0);
        __builtin_amdgcn_sched_group_barrier(SGB_MFMA_WMMA,  4, 0);
    }

#pragma unroll
    for (int t = 0; t < 4; ++t) {
        const int n     = nwave + t * 16 + lane16;
        const float bn  = bias[n];
        const int which = n / EMB;           // 0=q 1=k 2=v
        const int rem   = n % EMB;
        const int h     = rem / HDIM;
        const int d     = rem % HDIM;
        const float scale = (which == 0) ? SCALING : 1.0f;
#pragma unroll
        for (int j = 0; j < 8; ++j) {
            const int m = mbase + j + 8 * g;     // global row (b*S + s)
            const int b = m / SEQ;
            const int s = m % SEQ;
            const float val = (acc[t][j] + bn) * scale;
            const size_t bh = (size_t)b * HEADS + h;
            if (which == 2)
                VT[(bh * HDIM + d) * SEQ + s] = (__bf16)val;      // transposed
            else
                ((which == 0) ? Q : K)[(bh * SEQ + s) * HDIM + d] = (__bf16)val;
        }
    }
}

// =====================================================================
// Kernel 2: flash attention per head.  grid = (S/64, B*H), block = 128.
// K and V^T 64x64 tiles are staged into LDS by the Tensor Data Mover,
// double-buffered so the TDM copy of chunk kc+64 overlaps compute on kc.
// =====================================================================
__global__ __launch_bounds__(128)
void flash_attn_kernel(const __bf16* __restrict__ Q,    // [B*H, S, D]
                       const __bf16* __restrict__ K,    // [B*H, S, D]
                       const __bf16* __restrict__ VT,   // [B*H, D, S]
                       __bf16* __restrict__ AO) {       // [B*S, EMB]
    __shared__ __bf16 shK[2][64 * LDS_STRIDE];
    __shared__ __bf16 shV[2][64 * LDS_STRIDE];
    __shared__ __bf16 pbuf[4][16][72];                  // per-wave P tile

    const int lane   = threadIdx.x & 31;
    const int wave   = threadIdx.x >> 5;
    const int lane16 = lane & 15;
    const int g      = lane >> 4;

    const int bh = blockIdx.y;
    const int b  = bh / HEADS;
    const int h  = bh % HEADS;
    const int q0 = blockIdx.x * 64 + wave * 16;

    const __bf16* Qh  = Q  + (size_t)bh * SEQ * HDIM;
    const __bf16* Kh  = K  + (size_t)bh * SEQ * HDIM;
    const __bf16* VTh = VT + (size_t)bh * HDIM * SEQ;

    // Q fragments for this wave's 16 rows (D=64 -> two K=32 steps), loaded once.
    v16bf qa[2];
#pragma unroll
    for (int kk = 0; kk < 2; ++kk)
        qa[kk] = load_frag_bf16(Qh, HDIM, q0 + lane16, 32 * kk, g);

    v8f o[4];
    float rowm[8], rowl[8];
#pragma unroll
    for (int t = 0; t < 4; ++t) o[t] = (v8f)(0.0f);
#pragma unroll
    for (int j = 0; j < 8; ++j) { rowm[j] = -__builtin_inff(); rowl[j] = 0.0f; }

    // Prologue: TDM-load first K / V^T tiles into buffer 0.
    if (wave == 0) {
        tdm_load_2d(lds_off_of(&shK[0][0]), Kh, HDIM);           // K rows, d contiguous
        tdm_load_2d(lds_off_of(&shV[0][0]), VTh, SEQ);           // V^T rows (d), s contiguous
    }

    for (int kc = 0; kc < SEQ; kc += 64) {
        const int cur = (kc >> 6) & 1;
        if (wave == 0) __builtin_amdgcn_s_wait_tensorcnt(0);
        __syncthreads();                       // tile `cur` visible to all waves
        if (wave == 0 && (kc + 64) < SEQ) {    // prefetch next chunk (overlaps compute)
            tdm_load_2d(lds_off_of(&shK[cur ^ 1][0]), Kh + (size_t)(kc + 64) * HDIM, HDIM);
            tdm_load_2d(lds_off_of(&shV[cur ^ 1][0]), VTh + (kc + 64), SEQ);
        }
        const __bf16* Kt = &shK[cur][0];
        const __bf16* Vt = &shV[cur][0];

        // ---- S tile: scores[16 x 64] = Q(16x64) @ K^T ----
        v8f st[4];
#pragma unroll
        for (int t = 0; t < 4; ++t) {
            st[t] = (v8f)(0.0f);
#pragma unroll
            for (int kk = 0; kk < 2; ++kk) {
                v16bf kb = load_frag_bf16(Kt, LDS_STRIDE, t * 16 + lane16, 32 * kk, g);
                st[t] = WMMA_BF16(qa[kk], kb, st[t]);
            }
        }

        // ---- online softmax (row r = j + 8g lives across the 16 lanes of group g) ----
#pragma unroll
        for (int j = 0; j < 8; ++j) {
            float rm = fmaxf(fmaxf(st[0][j], st[1][j]), fmaxf(st[2][j], st[3][j]));
#pragma unroll
            for (int msk = 1; msk < 16; msk <<= 1)
                rm = fmaxf(rm, __shfl_xor(rm, msk, 32));
            const float mnew  = fmaxf(rowm[j], rm);
            const float alpha = __expf(rowm[j] - mnew);
            rowm[j] = mnew;
            float rs = 0.0f;
#pragma unroll
            for (int t = 0; t < 4; ++t) {
                const float p = __expf(st[t][j] - mnew);
                st[t][j] = p;
                rs += p;
            }
#pragma unroll
            for (int msk = 1; msk < 16; msk <<= 1)
                rs += __shfl_xor(rs, msk, 32);
            rowl[j] = rowl[j] * alpha + rs;
#pragma unroll
            for (int t = 0; t < 4; ++t) o[t][j] *= alpha;
        }

        // ---- C-layout -> A-layout conversion via LDS ----
        __syncthreads();
#pragma unroll
        for (int t = 0; t < 4; ++t)
#pragma unroll
            for (int j = 0; j < 8; ++j)
                pbuf[wave][j + 8 * g][t * 16 + lane16] = (__bf16)st[t][j];
        __syncthreads();

        // ---- O += P(16x64) @ V(64x64); V^T tile gives k-contiguous B-frags ----
#pragma unroll
        for (int kk = 0; kk < 2; ++kk) {
            v16bf pa = load_frag_bf16(&pbuf[wave][0][0], 72, lane16, 32 * kk, g);
#pragma unroll
            for (int t = 0; t < 4; ++t) {
                v16bf vb = load_frag_bf16(Vt, LDS_STRIDE, t * 16 + lane16, 32 * kk, g);
                o[t] = WMMA_BF16(pa, vb, o[t]);
            }
        }
    }

    // ---- normalize and write out ----
#pragma unroll
    for (int j = 0; j < 8; ++j) {
        const float inv = 1.0f / rowl[j];
        const int s = q0 + j + 8 * g;
        const size_t m = (size_t)b * SEQ + s;
#pragma unroll
        for (int t = 0; t < 4; ++t)
            AO[m * EMB + h * HDIM + t * 16 + lane16] = (__bf16)(o[t][j] * inv);
    }
}

// =====================================================================
// Kernel 3: out = AO @ Wout^T + bout  (fp32 result)
// =====================================================================
__global__ __launch_bounds__(128)
void out_proj_kernel(const __bf16* __restrict__ ao,    // [MTOT, EMB]
                     const float* __restrict__ w,      // [EMB, EMB]
                     const float* __restrict__ bias,   // [EMB]
                     float* __restrict__ out) {        // [MTOT, EMB]
    const int lane   = threadIdx.x & 31;
    const int wave   = threadIdx.x >> 5;
    const int lane16 = lane & 15;
    const int g      = lane >> 4;

    const int mbase = blockIdx.y * 16;
    const int nwave = blockIdx.x * 256 + wave * 64;

    v8f acc[4];
#pragma unroll
    for (int t = 0; t < 4; ++t) acc[t] = (v8f)(0.0f);

    const int arow = mbase + lane16;
    for (int kb = 0; kb < EMB; kb += 32) {
        // phase 1: issue ALL loads for this k-step
        rawfrag rb[4];
        v16bf a = load_frag_bf16(ao, EMB, arow, kb, g);
#pragma unroll
        for (int t = 0; t < 4; ++t)
            load_raw_f32(rb[t], w, EMB, nwave + t * 16 + lane16, kb, g);
        // phase 2: convert + matrix ops
#pragma unroll
        for (int t = 0; t < 4; ++t)
            acc[t] = WMMA_BF16(a, cvt_frag(rb[t]), acc[t]);
        // force pipeline: loads (A ~2 + B 16 reads), then cvts, then WMMAs
        __builtin_amdgcn_sched_group_barrier(SGB_VMEM_READ, 18, 0);
        __builtin_amdgcn_sched_group_barrier(SGB_VALU,      32, 0);
        __builtin_amdgcn_sched_group_barrier(SGB_MFMA_WMMA,  4, 0);
    }

#pragma unroll
    for (int t = 0; t < 4; ++t) {
        const int n = nwave + t * 16 + lane16;
        const float bn = bias[n];
#pragma unroll
        for (int j = 0; j < 8; ++j) {
            const int m = mbase + j + 8 * g;
            out[(size_t)m * EMB + n] = acc[t][j] + bn;
        }
    }
}

// =====================================================================
extern "C" void kernel_launch(void* const* d_in, const int* in_sizes, int n_in,
                              void* d_out, int out_size, void* d_ws, size_t ws_size,
                              hipStream_t stream) {
    const float* x    = (const float*)d_in[0];
    const float* wqkv = (const float*)d_in[1];
    const float* bqkv = (const float*)d_in[2];
    const float* wout = (const float*)d_in[3];
    const float* bout = (const float*)d_in[4];
    float* out = (float*)d_out;

    const size_t elems = (size_t)BATCH * HEADS * SEQ * HDIM;   // 6,291,456
    __bf16* Q  = (__bf16*)d_ws;
    __bf16* K  = Q + elems;
    __bf16* VT = K + elems;                                    // [B*H, D, S]
    __bf16* AO = VT + elems;                                   // [MTOT, EMB]

    qkv_proj_kernel<<<dim3(QKV_N / 256, MTOT / 16), 128, 0, stream>>>(x, wqkv, bqkv, Q, K, VT);
    flash_attn_kernel<<<dim3(SEQ / 64, BATCH * HEADS), 128, 0, stream>>>(Q, K, VT, AO);
    out_proj_kernel<<<dim3(EMB / 256, MTOT / 16), 128, 0, stream>>>(AO, wout, bout, out);
}
